// REALS_model_62105227100309
// MI455X (gfx1250) — compile-verified, matrix operand order checked by reading, not run on hardware
//
#include <hip/hip_runtime.h>
#include <hip/hip_bf16.h>

// Problem constants (match reference)
#define T_FRAMES 128
#define H_IMG 512
#define W_IMG 512
#define INP (H_IMG * W_IMG)          // 262144
#define TILE_PX 2048                  // pixels per block in warp kernel
#define TILES_PER_FRAME (INP / TILE_PX)  // 128

typedef __attribute__((ext_vector_type(16))) _Float16 v16h;
typedef __attribute__((ext_vector_type(8)))  float    v8f;
typedef __attribute__((ext_vector_type(2)))  float    v2f;

// ---------------------------------------------------------------------------
// Kernel 1: affine warp + bilinear sample (zero padding, align_corners=False)
// Writes x_reg and per-block partial dot products  partial[f*128 + tile].
// ---------------------------------------------------------------------------
__global__ __launch_bounds__(256)
void k_warp(const float* __restrict__ x,
            const float* __restrict__ theta_ro,
            const float* __restrict__ theta_tr,
            const float* __restrict__ Wv,
            float* __restrict__ x_reg,
            float* __restrict__ partial)
{
    const int tile = blockIdx.x;          // 0..127 within frame
    const int f    = blockIdx.y;          // frame 0..127
    const int tid  = threadIdx.x;         // 0..255
    const int lane = tid & 31;            // wave32
    const int wid  = tid >> 5;            // 8 waves / block

    // per-frame affine params (uniform across block)
    const float ang = theta_ro[f];
    const float c = cosf(ang), s = sinf(ang);
    const float tx = theta_tr[2 * f + 0];
    const float ty = theta_tr[2 * f + 1];

    const float* __restrict__ img = x + (size_t)f * INP;
    const int tile_base = tile * TILE_PX;

    float acc = 0.f;

#pragma unroll
    for (int i = 0; i < TILE_PX / 256; ++i) {
        const int p = tile_base + tid + i * 256;   // coalesced

        // prefetch W for the next chunk (gfx1250 global_prefetch_b8)
        __builtin_prefetch(Wv + p + 256, 0, 0);

        const int py = p >> 9;
        const int px = p & (W_IMG - 1);

        const float gx = (2.f * (float)px + 1.f) * (1.f / (float)W_IMG) - 1.f;
        const float gy = (2.f * (float)py + 1.f) * (1.f / (float)H_IMG) - 1.f;

        const float ox = c * gx - s * gy + tx;
        const float oy = s * gx + c * gy + ty;

        const float ix = ((ox + 1.f) * (float)W_IMG - 1.f) * 0.5f;
        const float iy = ((oy + 1.f) * (float)H_IMG - 1.f) * 0.5f;

        const float x0f = floorf(ix);
        const float y0f = floorf(iy);
        const int   x0  = (int)x0f;
        const int   y0  = (int)y0f;
        const float wx1 = ix - x0f, wx0 = 1.f - wx1;
        const float wy1 = iy - y0f, wy0 = 1.f - wy1;

        // zero-padded gathers
        float v00 = 0.f, v01 = 0.f, v10 = 0.f, v11 = 0.f;
        const bool xv0 = (x0 >= 0) & (x0 <= W_IMG - 1);
        const bool xv1 = (x0 + 1 >= 0) & (x0 + 1 <= W_IMG - 1);
        const bool yv0 = (y0 >= 0) & (y0 <= H_IMG - 1);
        const bool yv1 = (y0 + 1 >= 0) & (y0 + 1 <= H_IMG - 1);
        if (yv0) {
            const float* row = img + ((size_t)y0 << 9);
            if (xv0) v00 = row[x0];
            if (xv1) v01 = row[x0 + 1];
        }
        if (yv1) {
            const float* row = img + ((size_t)(y0 + 1) << 9);
            if (xv0) v10 = row[x0];
            if (xv1) v11 = row[x0 + 1];
        }

        const float val = wy0 * (wx0 * v00 + wx1 * v01)
                        + wy1 * (wx0 * v10 + wx1 * v11);

        x_reg[(size_t)f * INP + p] = val;
        acc += val * Wv[p];
    }

    // wave32 reduction
#pragma unroll
    for (int off = 16; off > 0; off >>= 1)
        acc += __shfl_down(acc, off, 32);

    __shared__ float red[8];
    if (lane == 0) red[wid] = acc;
    __syncthreads();
    if (tid == 0) {
        float sum = 0.f;
#pragma unroll
        for (int w = 0; w < 8; ++w) sum += red[w];
        partial[f * TILES_PER_FRAME + tile] = sum;  // deterministic (no atomics)
    }
}

// ---------------------------------------------------------------------------
// Kernel 2: reduce 128 partials per frame -> s[f]  (deterministic tree)
// ---------------------------------------------------------------------------
__global__ __launch_bounds__(128)
void k_dot(const float* __restrict__ partial, float* __restrict__ s_out)
{
    const int f    = blockIdx.x;
    const int tid  = threadIdx.x;   // 0..127
    const int lane = tid & 31;
    const int wid  = tid >> 5;      // 4 waves

    float v = partial[f * TILES_PER_FRAME + tid];
#pragma unroll
    for (int off = 16; off > 0; off >>= 1)
        v += __shfl_down(v, off, 32);

    __shared__ float red[4];
    if (lane == 0) red[wid] = v;
    __syncthreads();
    if (tid == 0)
        s_out[f] = red[0] + red[1] + red[2] + red[3];
}

// ---------------------------------------------------------------------------
// Kernel 3: rank-1 outer product  L[t, i] = s[t] * W[i]  via WMMA.
// Preferred: V_WMMA_F32_16X16X4_F32 (exact fp32): D = A(16x4) x B(4x16) + 0
// with A col K=0 = s-group, B row K=0 = W-chunk, other K zero.
// Fallback (if builtin undeclared): f16 WMMA as in round 1.
// One wave = one 16-frame group; 16 chunks of 16 pixels per wave, fully
// unrolled so the 16 coalesced W loads clause together and overlap WMMAs.
// ---------------------------------------------------------------------------
__global__ __launch_bounds__(256)
void k_outer(const float* __restrict__ s_vec,
             const float* __restrict__ Wv,
             float* __restrict__ L)
{
    const int tid  = threadIdx.x;
    const int lane = tid & 31;
    const int fg   = tid >> 5;                 // 0..7 -> frame group (16 frames)
    const int chunk_block = blockIdx.x;        // 0..1023, 256 pixels each

    const int l15  = lane & 15;
    const bool lo  = (lane < 16);

    // s for this frame group: branchless, coalesced (lanes 16-31 duplicate then
    // select-to-zero -> v_cndmask, no exec-mask branching).
    const float sv = s_vec[fg * 16 + l15];

#if __has_builtin(__builtin_amdgcn_wmma_f32_16x16x4_f32)
    // A matrix 16x4 f32 (2 VGPRs): lane m (0..15): VGPR0=K0, VGPR1=K1;
    // lanes 16..31: K2, K3. Only K0 nonzero -> exact fp32 rank-1 product.
    v2f a;
    a[0] = lo ? sv : 0.f;
    a[1] = 0.f;
#else
    // Fallback: f16 A (16x32), K=0 only.
    v16h a;
#pragma unroll
    for (int i = 0; i < 16; ++i) a[i] = (_Float16)0.f;
    a[0] = lo ? (_Float16)sv : (_Float16)0.f;
#endif

    const size_t row_stride = (size_t)INP;
    const int m_off = (lane >> 4) * 8;   // C/D layout: lanes 16-31 -> M+8

#pragma unroll
    for (int ch = 0; ch < 16; ++ch) {
        const int n_base = chunk_block * 256 + ch * 16;

        // W chunk: unconditional coalesced load from clamped lane address.
        const float wv = Wv[n_base + l15];

#if __has_builtin(__builtin_amdgcn_wmma_f32_16x16x4_f32)
        // B matrix 4x16 f32 (2 VGPRs): lane n (0..15): VGPR0=K0, VGPR1=K1.
        v2f b;
        b[0] = lo ? wv : 0.f;
        b[1] = 0.f;

        v8f cacc = {};
        cacc = __builtin_amdgcn_wmma_f32_16x16x4_f32(
            /*neg_a=*/false, a, /*neg_b=*/false, b,
            /*c_mod=*/(short)0, cacc, /*reuse_a=*/false, /*reuse_b=*/false);
#else
        v16h b;
#pragma unroll
        for (int i = 0; i < 16; ++i) b[i] = (_Float16)0.f;
        b[0] = lo ? (_Float16)wv : (_Float16)0.f;

        v8f cacc = {};
        cacc = __builtin_amdgcn_wmma_f32_16x16x32_f16(
            false, a, false, b, (short)0, cacc, false, false);
#endif

        // C/D layout: VGPR r -> M=r (lanes 0-15), M=r+8 (lanes 16-31); N=lane%16
        const int n = n_base + l15;
#pragma unroll
        for (int r = 0; r < 8; ++r) {
            const int row = fg * 16 + r + m_off;
            L[(size_t)row * row_stride + n] = cacc[r];
        }
    }
}

// ---------------------------------------------------------------------------
extern "C" void kernel_launch(void* const* d_in, const int* in_sizes, int n_in,
                              void* d_out, int out_size, void* d_ws, size_t ws_size,
                              hipStream_t stream)
{
    const float* x        = (const float*)d_in[0];   // (128, 262144)
    const float* theta_ro = (const float*)d_in[1];   // (128, 1)
    const float* theta_tr = (const float*)d_in[2];   // (128, 2)
    const float* Wv       = (const float*)d_in[3];   // (1, 262144)

    float* out   = (float*)d_out;                    // [x_reg | L]
    float* x_reg = out;
    float* L     = out + (size_t)T_FRAMES * INP;

    float* ws      = (float*)d_ws;
    float* partial = ws;                              // 128*128 floats
    float* s_vec   = ws + T_FRAMES * TILES_PER_FRAME; // 128 floats

    // 1) warp + bilinear + per-block dot partials
    dim3 g1(TILES_PER_FRAME, T_FRAMES);
    k_warp<<<g1, 256, 0, stream>>>(x, theta_ro, theta_tr, Wv, x_reg, partial);

    // 2) deterministic reduction -> s
    k_dot<<<T_FRAMES, 128, 0, stream>>>(partial, s_vec);

    // 3) L = s ⊗ W via WMMA (8 frame-groups x 1024 chunk-blocks of 256 px)
    k_outer<<<INP / 256, 256, 0, stream>>>(s_vec, Wv, L);
}
